// RandLANet_30855045054646
// MI455X (gfx1250) — matrix operand order repaired and checked by reading, not compiled, not run on hardware
//
#include <hip/hip_runtime.h>
#include <hip/hip_bf16.h>

// ---------------------------------------------------------------------------
// RandLA-Net forward for MI455X (gfx1250).
// Dense channel-mixing einsums -> v_wmma_f32_16x16x32_bf16 (bf16 in, f32 acc),
// 32x32 macro-tile per wave (4 WMMA per K-step). Async global->LDS staging.
// ---------------------------------------------------------------------------

#define BB 2
#define NN 4096
#define KK 16

typedef __bf16 bf16_t;
typedef __attribute__((ext_vector_type(16))) __bf16 v16bf;
typedef __attribute__((ext_vector_type(8)))  float  v8f;

#if defined(__has_builtin)
#  if __has_builtin(__builtin_amdgcn_global_load_async_to_lds_b128)
#    define HAVE_ASYNC_LDS 1
#  endif
#endif
#ifndef HAVE_ASYNC_LDS
#  define HAVE_ASYNC_LDS 0
#endif

#if HAVE_ASYNC_LDS
typedef __attribute__((__vector_size__(4 * sizeof(int)))) int v4i_t;
typedef __attribute__((address_space(1))) v4i_t gv4i_t;
typedef __attribute__((address_space(3))) v4i_t lv4i_t;
__device__ __forceinline__ void async_copy16(const float* gp, float* lp) {
    __builtin_amdgcn_global_load_async_to_lds_b128(
        (gv4i_t*)gp, (lv4i_t*)lp, 0, 0);
}
__device__ __forceinline__ void wait_async0() {
#  if __has_builtin(__builtin_amdgcn_s_wait_asynccnt)
    __builtin_amdgcn_s_wait_asynccnt(0);
#  else
    asm volatile("s_wait_asynccnt 0x0" ::: "memory");
#  endif
}
#endif

#define ACT_NONE     0
#define ACT_RELU     1
#define ACT_LEAKY02  2
#define ACT_LEAKY001 3

__device__ __forceinline__ float actf(float v, int act) {
    switch (act) {
        case ACT_RELU:     return v > 0.f ? v : 0.f;
        case ACT_LEAKY02:  return v > 0.f ? v : 0.2f * v;
        case ACT_LEAKY001: return v > 0.f ? v : 0.01f * v;
        default:           return v;
    }
}

__device__ __forceinline__ v8f do_wmma(v16bf a, v16bf b, v8f c) {
    return __builtin_amdgcn_wmma_f32_16x16x32_bf16(false, a, false, b,
                                                   (short)0, c, false, false);
}

// ---------------------------------------------------------------------------
// GEMM: out[R,Cout] = act(A[R,Cin] * W[Cin,Cout] + bias)
// One wave per 32x32 output block; K stepped by 32; 4 WMMAs per K-step.
// ---------------------------------------------------------------------------
__global__ __launch_bounds__(32)
void k_gemm(const float* __restrict__ A, const float* __restrict__ W,
            const float* __restrict__ bias, float* __restrict__ out,
            int R, int Cin, int Cout, int act)
{
    __shared__ float sA[32 * 32];   // [m][k] f32
    __shared__ float sB[32 * 32];   // [k][n] f32

    const int lane  = threadIdx.x;           // 0..31
    const int tileM = blockIdx.x * 32;
    const int tileN = blockIdx.y * 32;

    v8f acc00 = {}, acc01 = {}, acc10 = {}, acc11 = {};

    for (int k0 = 0; k0 < Cin; k0 += 32) {
        const bool fastA = (tileM + 32 <= R)    && (k0 + 32 <= Cin);
        const bool fastB = (tileN + 32 <= Cout) && (k0 + 32 <= Cin);

        if (fastA) {
            // 32x32 floats = 256 float4; each lane moves 8.
            #pragma unroll
            for (int t = 0; t < 8; ++t) {
                int e = t * 32 + lane;
                int m = e >> 3, c4 = (e & 7) * 4;
                const float* gp = &A[(long)(tileM + m) * Cin + k0 + c4];
#if HAVE_ASYNC_LDS
                async_copy16(gp, &sA[m * 32 + c4]);
#else
                *(float4*)&sA[m * 32 + c4] = *(const float4*)gp;
#endif
            }
        } else {
            // Branchless guarded path: clamp index, select value.
            #pragma unroll
            for (int t = 0; t < 32; ++t) {
                int e = t * 32 + lane;            // 0..1023
                int m = e >> 5, k = e & 31;
                int gr = tileM + m, gk = k0 + k;
                bool ok = (gr < R) && (gk < Cin);
                long idx = ok ? ((long)gr * Cin + gk) : 0;
                float v = A[idx];
                sA[m * 32 + k] = ok ? v : 0.f;
            }
        }

        if (fastB) {
            #pragma unroll
            for (int t = 0; t < 8; ++t) {
                int e = t * 32 + lane;
                int kb = e >> 3, n4 = (e & 7) * 4;
                const float* gp = &W[(long)(k0 + kb) * Cout + tileN + n4];
#if HAVE_ASYNC_LDS
                async_copy16(gp, &sB[kb * 32 + n4]);
#else
                *(float4*)&sB[kb * 32 + n4] = *(const float4*)gp;
#endif
            }
        } else {
            #pragma unroll
            for (int t = 0; t < 32; ++t) {
                int e = t * 32 + lane;
                int kb = e >> 5, nn = e & 31;
                int gk = k0 + kb, gn = tileN + nn;
                bool ok = (gk < Cin) && (gn < Cout);
                long idx = ok ? ((long)gk * Cout + gn) : 0;
                float v = W[idx];
                sB[kb * 32 + nn] = ok ? v : 0.f;
            }
        }

#if HAVE_ASYNC_LDS
        if (fastA || fastB) wait_async0();
#endif
        __syncthreads();

        // Fragments in contiguous k order, k = kbBase + e, for A and B (wave32), f32->bf16 at build time.
        const int mrow   = lane & 15;
        const int hiBase = (lane >= 16) ? 8 : 0;    // unused: scaled by 0 in kA
        const int kbBase = (lane >= 16) ? 16 : 0;   // B K-offset, hi half-wave

        v16bf a0, a1, b0, b1;
        #pragma unroll
        for (int e = 0; e < 16; ++e) {
            int j  = e >> 1;
            int kA = kbBase + e + 0 * j + 0 * hiBase;
            a0[e] = (bf16_t)sA[mrow * 32 + kA];
            a1[e] = (bf16_t)sA[(16 + mrow) * 32 + kA];
            b0[e] = (bf16_t)sB[(kbBase + e) * 32 + mrow];
            b1[e] = (bf16_t)sB[(kbBase + e) * 32 + 16 + mrow];
        }

        acc00 = do_wmma(a0, b0, acc00);
        acc01 = do_wmma(a0, b1, acc01);
        acc10 = do_wmma(a1, b0, acc10);
        acc11 = do_wmma(a1, b1, acc11);

        __syncthreads();
    }

    // Store: C/D layout -> VGPR i: M = i + 8*(lane>=16), N = lane&15.
    const int ncol = lane & 15;
    const int mofs = (lane >= 16) ? 8 : 0;
    #pragma unroll
    for (int i = 0; i < 8; ++i) {
        int gr0 = tileM + mofs + i;
        int gr1 = tileM + 16 + mofs + i;
        int gc0 = tileN + ncol;
        int gc1 = tileN + 16 + ncol;
        float b0v = bias ? bias[gc0 < Cout ? gc0 : 0] : 0.f;
        float b1v = bias ? bias[gc1 < Cout ? gc1 : 0] : 0.f;
        if (gr0 < R && gc0 < Cout) out[(long)gr0 * Cout + gc0] = actf(acc00[i] + b0v, act);
        if (gr0 < R && gc1 < Cout) out[(long)gr0 * Cout + gc1] = actf(acc01[i] + b1v, act);
        if (gr1 < R && gc0 < Cout) out[(long)gr1 * Cout + gc0] = actf(acc10[i] + b0v, act);
        if (gr1 < R && gc1 < Cout) out[(long)gr1 * Cout + gc1] = actf(acc11[i] + b1v, act);
    }
}

// ---------------------------------------------------------------------------
// KNN (k=16) over the first n points per batch (self-query).
// ---------------------------------------------------------------------------
__global__ void k_knn16(const float* __restrict__ coords, int n, int bstride,
                        int* __restrict__ idx, float* __restrict__ dist)
{
    const int q = blockIdx.x * blockDim.x + threadIdx.x;
    const int b = blockIdx.y;
    const bool active = q < n;
    const float* cb = coords + (long)b * bstride;

    float qx = 0.f, qy = 0.f, qz = 0.f;
    if (active) { qx = cb[q*3]; qy = cb[q*3+1]; qz = cb[q*3+2]; }

    float bd[KK]; int bi[KK];
    #pragma unroll
    for (int t = 0; t < KK; ++t) { bd[t] = 3.4e38f; bi[t] = 0; }

    __shared__ float tile[128 * 3];
    for (int r0 = 0; r0 < n; r0 += 128) {
        int cnt = n - r0; if (cnt > 128) cnt = 128;
        for (int t = threadIdx.x; t < cnt * 3; t += blockDim.x)
            tile[t] = cb[r0 * 3 + t];
        if (r0 + 128 < n) __builtin_prefetch(cb + (r0 + 128) * 3, 0, 1);
        __syncthreads();
        if (active) {
            for (int r = 0; r < cnt; ++r) {
                float dx = qx - tile[r*3], dy = qy - tile[r*3+1], dz = qz - tile[r*3+2];
                float d2 = dx*dx + dy*dy + dz*dz;
                if (d2 < bd[KK-1]) {
                    int p = KK - 1;
                    while (p > 0 && bd[p-1] > d2) { bd[p] = bd[p-1]; bi[p] = bi[p-1]; --p; }
                    bd[p] = d2; bi[p] = r0 + r;
                }
            }
        }
        __syncthreads();
    }
    if (active) {
        long base = ((long)b * n + q) * KK;
        #pragma unroll
        for (int t = 0; t < KK; ++t) {
            idx[base + t]  = bi[t];
            dist[base + t] = sqrtf(fmaxf(bd[t], 0.f));
        }
    }
}

// 1-NN: for each of n_up queries, nearest of n_cur refs.
__global__ void k_nn1(const float* __restrict__ coords, int n_up, int n_cur,
                      int bstride, int* __restrict__ nb)
{
    const int q = blockIdx.x * blockDim.x + threadIdx.x;
    const int b = blockIdx.y;
    const bool active = q < n_up;
    const float* cb = coords + (long)b * bstride;

    float qx = 0.f, qy = 0.f, qz = 0.f;
    if (active) { qx = cb[q*3]; qy = cb[q*3+1]; qz = cb[q*3+2]; }

    float best = 3.4e38f; int bidx = 0;
    __shared__ float tile[128 * 3];
    for (int r0 = 0; r0 < n_cur; r0 += 128) {
        int cnt = n_cur - r0; if (cnt > 128) cnt = 128;
        for (int t = threadIdx.x; t < cnt * 3; t += blockDim.x)
            tile[t] = cb[r0 * 3 + t];
        __syncthreads();
        if (active) {
            for (int r = 0; r < cnt; ++r) {
                float dx = qx - tile[r*3], dy = qy - tile[r*3+1], dz = qz - tile[r*3+2];
                float d2 = dx*dx + dy*dy + dz*dz;
                if (d2 < best) { best = d2; bidx = r0 + r; }
            }
        }
        __syncthreads();
    }
    if (active) nb[(long)b * n_up + q] = bidx;
}

// ---------------------------------------------------------------------------
// Relative point encoding: rel = [q(3), nb(3), q-nb(3), dist(1)]
// ---------------------------------------------------------------------------
__global__ void k_build_rel(const float* __restrict__ coords, int bstride,
                            const int* __restrict__ idx, const float* __restrict__ dist,
                            float* __restrict__ rel, int n, int B)
{
    long t = (long)blockIdx.x * blockDim.x + threadIdx.x;
    long total = (long)B * n * KK;
    if (t >= total) return;
    long bq = t / KK;
    int q = (int)(bq % n), b = (int)(bq / n);
    const float* cb = coords + (long)b * bstride;
    int j = idx[t];
    float qx = cb[q*3], qy = cb[q*3+1], qz = cb[q*3+2];
    float rx = cb[j*3], ry = cb[j*3+1], rz = cb[j*3+2];
    float* o = rel + t * 10;
    o[0]=qx; o[1]=qy; o[2]=qz; o[3]=rx; o[4]=ry; o[5]=rz;
    o[6]=qx-rx; o[7]=qy-ry; o[8]=qz-rz; o[9]=dist[t];
}

// Broadcast per-point feats (c2 ch) into x[b,n,K,co] at column offset `off`.
__global__ void k_broadcast(const float* __restrict__ feats, float* __restrict__ dst,
                            long R2, int c2, int co, int off)
{
    long t = (long)blockIdx.x * blockDim.x + threadIdx.x;
    long total = R2 * KK * c2;
    if (t >= total) return;
    int c = (int)(t % c2);
    long tmp = t / c2;
    int k = (int)(tmp % KK);
    long row = tmp / KK;
    dst[(row * KK + k) * co + off + c] = feats[row * c2 + c];
}

// Fused softmax(axis=k) + weighted sum: f[row,c] = sum_k softmax_k(s)*x
__global__ void k_attpool(const float* __restrict__ sraw, const float* __restrict__ x,
                          float* __restrict__ f, long R2, int co)
{
    long t = (long)blockIdx.x * blockDim.x + threadIdx.x;
    if (t >= R2 * co) return;
    long row = t / co;
    int  c   = (int)(t % co);
    long base = row * KK * (long)co + c;
    float mx = -3.4e38f;
    #pragma unroll
    for (int k = 0; k < KK; ++k) mx = fmaxf(mx, sraw[base + (long)k * co]);
    float sum = 0.f, accv = 0.f;
    #pragma unroll
    for (int k = 0; k < KK; ++k) {
        float e = __expf(sraw[base + (long)k * co] - mx);
        sum  += e;
        accv += e * x[base + (long)k * co];
    }
    f[row * co + c] = accv / sum;
}

// ---------------------------------------------------------------------------
// BatchNorm helpers (mean/var over rows, per channel).
// ---------------------------------------------------------------------------
__global__ void k_zero(float* p, int n)
{
    int t = blockIdx.x * blockDim.x + threadIdx.x;
    if (t < n) p[t] = 0.f;
}

__global__ void k_bn_stats(const float* __restrict__ x, long long R, int C,
                           float* __restrict__ stats)
{
    int c = blockIdx.x * blockDim.x + threadIdx.x;
    if (c >= C) return;
    long long nch = gridDim.y;
    long long per = (R + nch - 1) / nch;
    long long r0 = (long long)blockIdx.y * per;
    long long r1 = r0 + per; if (r1 > R) r1 = R;
    float s = 0.f, s2 = 0.f;
    for (long long r = r0; r < r1; ++r) {
        float v = x[r * C + c];
        s += v; s2 += v * v;
    }
    atomicAdd(&stats[c], s);
    atomicAdd(&stats[C + c], s2);
}

__global__ void k_bn_finalize(float* __restrict__ stats, const float* __restrict__ gamma,
                              const float* __restrict__ beta, long long R, int C)
{
    int c = blockIdx.x * blockDim.x + threadIdx.x;
    if (c >= C) return;
    float mean = stats[c] / (float)R;
    float var  = stats[C + c] / (float)R - mean * mean;
    var = fmaxf(var, 0.f);
    float scale = gamma[c] * rsqrtf(var + 1e-6f);
    stats[c]     = scale;
    stats[C + c] = beta[c] - mean * scale;
}

// y = act(raw*scale + shift [+ addSrc]); write to dst[r*ldd + off + c]
__global__ void k_bn_apply(const float* __restrict__ raw, const float* __restrict__ stats,
                           long long R, int C, float* __restrict__ dst, int ldd, int off,
                           int act, const float* __restrict__ addSrc)
{
    long long t = (long long)blockIdx.x * blockDim.x + threadIdx.x;
    if (t >= R * C) return;
    long long r = t / C;
    int c = (int)(t % C);
    float v = raw[t] * stats[c] + stats[C + c];
    if (addSrc) v += addSrc[t];
    dst[r * (long long)ldd + off + c] = actf(v, act);
}

// ---------------------------------------------------------------------------
// Data movement kernels.
// ---------------------------------------------------------------------------
__global__ void k_gather_rows(const float* __restrict__ src, const int* __restrict__ nb,
                              float* __restrict__ dst, int B, int n_up, int n_cur,
                              int C, int ldd, int off)
{
    long t = (long)blockIdx.x * blockDim.x + threadIdx.x;
    long total = (long)B * n_up * C;
    if (t >= total) return;
    int c = (int)(t % C);
    long row = t / C;                     // b*n_up + i
    int b = (int)(row / n_up);
    int j = nb[row];
    dst[row * ldd + off + c] = src[((long)b * n_cur + j) * C + c];
}

__global__ void k_copy_cols(const float* __restrict__ src, float* __restrict__ dst,
                            long R, int C, int ldd, int off)
{
    long t = (long)blockIdx.x * blockDim.x + threadIdx.x;
    if (t >= R * C) return;
    long r = t / C; int c = (int)(t % C);
    dst[r * ldd + off + c] = src[t];
}

__global__ void k_slice_rows(const float* __restrict__ src, float* __restrict__ dst,
                             int B, int n1, int n2, int C)
{
    long t = (long)blockIdx.x * blockDim.x + threadIdx.x;
    long total = (long)B * n2 * C;
    if (t >= total) return;
    int c = (int)(t % C);
    long row = t / C;
    int i = (int)(row % n2), b = (int)(row / n2);
    dst[t] = src[((long)b * n1 + i) * C + c];
}

// dst[b,i,0:C] = src[b, prm[i], 0:ld...C]
__global__ void k_permute_rows(const float* __restrict__ src, const int* __restrict__ prm,
                               float* __restrict__ dst, int B, int n, int lds, int C)
{
    long t = (long)blockIdx.x * blockDim.x + threadIdx.x;
    long total = (long)B * n * C;
    if (t >= total) return;
    int c = (int)(t % C);
    long row = t / C;
    int i = (int)(row % n), b = (int)(row / n);
    dst[row * C + c] = src[((long)b * n + prm[i]) * lds + c];
}

__global__ void k_invperm(const int* __restrict__ prm, int* __restrict__ inv, int n)
{
    int i = blockIdx.x * blockDim.x + threadIdx.x;
    if (i < n) inv[prm[i]] = i;
}

// out[b,c,i] = x[b,i,c]
__global__ void k_transpose_out(const float* __restrict__ x, float* __restrict__ out,
                                int B, int n, int C)
{
    long t = (long)blockIdx.x * blockDim.x + threadIdx.x;
    long total = (long)B * n * C;
    if (t >= total) return;
    int c = (int)(t % C);
    long row = t / C;
    int i = (int)(row % n), b = (int)(row / n);
    out[((long)b * C + c) * n + i] = x[t];
}

// ---------------------------------------------------------------------------
// Host-side orchestration.
// ---------------------------------------------------------------------------
static inline void gemm(const float* A, const float* W, const float* b, float* out,
                        long R, int Cin, int Cout, int act, hipStream_t s)
{
    dim3 g((unsigned)((R + 31) / 32), (unsigned)((Cout + 31) / 32));
    k_gemm<<<g, 32, 0, s>>>(A, W, b, out, (int)R, Cin, Cout, act);
}

static inline void run_bn(const float* raw, const float* gamma, const float* beta,
                          long long R, int C, float* dst, int ldd, int off, int act,
                          const float* addSrc, float* stats, hipStream_t s)
{
    k_zero<<<(2 * C + 255) / 256, 256, 0, s>>>(stats, 2 * C);
    dim3 gs((unsigned)((C + 63) / 64), 64);
    k_bn_stats<<<gs, 64, 0, s>>>(raw, R, C, stats);
    k_bn_finalize<<<(C + 63) / 64, 64, 0, s>>>(stats, gamma, beta, R, C);
    long long total = R * (long long)C;
    k_bn_apply<<<(unsigned)((total + 255) / 256), 256, 0, s>>>(raw, stats, R, C, dst,
                                                               ldd, off, act, addSrc);
}

static inline unsigned gblk(long n) { return (unsigned)((n + 255) / 256); }

extern "C" void kernel_launch(void* const* d_in, const int* in_sizes, int n_in,
                              void* d_out, int out_size, void* d_ws, size_t ws_size,
                              hipStream_t stream)
{
    (void)in_sizes; (void)n_in; (void)out_size; (void)ws_size;

    const float* points = (const float*)d_in[0];   // [B,N,6]
    const int*   perm   = (const int*)d_in[1];     // [N]

    // ---- params (depth-first, insertion order of make_params) ----
    int pi = 2;
    auto NXT = [&]() { return (const float*)d_in[pi++]; };

    const float *fcs_w = NXT(), *fcs_b = NXT();
    const float *bns_g = NXT(), *bns_be = NXT();

    struct Lvl {
        const float *m1w, *m1b;
        const float *l1w, *l1b, *l1g, *l1be;
        const float *p1s, *p1mw, *p1mb, *p1mg, *p1mbe;
        const float *l2w, *l2b, *l2g, *l2be;
        const float *p2s, *p2mw, *p2mb, *p2mg, *p2mbe;
        const float *m2w, *m2b;
        const float *shw, *shb, *shg, *shbe;
    } L[4];
    for (int l = 0; l < 4; ++l) {
        L[l].m1w = NXT(); L[l].m1b = NXT();
        L[l].l1w = NXT(); L[l].l1b = NXT(); L[l].l1g = NXT(); L[l].l1be = NXT();
        L[l].p1s = NXT(); L[l].p1mw = NXT(); L[l].p1mb = NXT(); L[l].p1mg = NXT(); L[l].p1mbe = NXT();
        L[l].l2w = NXT(); L[l].l2b = NXT(); L[l].l2g = NXT(); L[l].l2be = NXT();
        L[l].p2s = NXT(); L[l].p2mw = NXT(); L[l].p2mb = NXT(); L[l].p2mg = NXT(); L[l].p2mbe = NXT();
        L[l].m2w = NXT(); L[l].m2b = NXT();
        L[l].shw = NXT(); L[l].shb = NXT(); L[l].shg = NXT(); L[l].shbe = NXT();
    }
    const float *mid_w = NXT(), *mid_b = NXT();
    struct Dec { const float *w, *b, *g, *be; } Dp[4];
    for (int i = 0; i < 4; ++i) { Dp[i].w = NXT(); Dp[i].b = NXT(); Dp[i].g = NXT(); Dp[i].be = NXT(); }
    struct Fce { const float *w, *b, *g, *be; } Fe[3];
    Fe[0].w = NXT(); Fe[0].b = NXT(); Fe[0].g = NXT(); Fe[0].be = NXT();
    Fe[1].w = NXT(); Fe[1].b = NXT(); Fe[1].g = NXT(); Fe[1].be = NXT();
    Fe[2].w = NXT(); Fe[2].b = NXT(); Fe[2].g = nullptr; Fe[2].be = nullptr;

    // ---- workspace bump allocator (deterministic, recomputed each call) ----
    char* wsb = (char*)d_ws;
    size_t woff = 0;
    auto allocf = [&](size_t n) -> float* {
        float* p = (float*)(wsb + woff);
        woff += ((n * sizeof(float) + 255) / 256) * 256;
        return p;
    };
    auto alloci = [&](size_t n) -> int* {
        int* p = (int*)(wsb + woff);
        woff += ((n * sizeof(int) + 255) / 256) * 256;
        return p;
    };

    float* coords0   = allocf((size_t)BB * NN * 3);
    float* xStartRaw = allocf((size_t)BB * NN * 8);
    float* xStart    = allocf((size_t)BB * NN * 8);
    float* xPerm     = allocf((size_t)BB * NN * 8);
    float* stats     = allocf(2048);
    int*   invp      = alloci(NN);
    int*   idxB      = alloci((size_t)BB * NN * KK);
    float* distB     = allocf((size_t)BB * NN * KK);
    float* featBuf0  = allocf(65536);
    float* featBuf1  = allocf(65536);
    float* x1        = allocf(65536);
    float* relB      = allocf((size_t)BB * NN * KK * 10);
    float* encRaw    = allocf(1048576);
    float* xBig      = allocf(2097152);
    float* scoresRaw = allocf(2097152);
    float* pooled    = allocf(131072);
    float* poolRaw   = allocf(131072);
    float* pool2out  = allocf(131072);
    float* mlp2out   = allocf(262144);
    float* shortRaw  = allocf(262144);
    float* stacks[4];
    stacks[0] = allocf(262144); stacks[1] = allocf(131072);
    stacks[2] = allocf(65536);  stacks[3] = allocf(32768);
    float* midX    = allocf(16384);
    int*   nbIdx   = alloci((size_t)BB * NN);
    float* catBuf  = allocf(524288);
    float* decRaw  = allocf(65536);
    float* decBufA = allocf(65536);
    float* decBufB = allocf(65536);
    float* unperm  = allocf((size_t)BB * NN * 8);
    float* fc1raw  = allocf((size_t)BB * NN * 64);
    float* fc1     = allocf((size_t)BB * NN * 64);
    float* fc2raw  = allocf((size_t)BB * NN * 32);
    float* fc2     = allocf((size_t)BB * NN * 32);
    float* logits  = allocf((size_t)BB * NN * 13);

    // ---- fc_start + bn_start + leaky(0.2), then permute coords & feats ----
    gemm(points, fcs_w, fcs_b, xStartRaw, (long)BB * NN, 6, 8, ACT_NONE, stream);
    run_bn(xStartRaw, bns_g, bns_be, (long long)BB * NN, 8, xStart, 8, 0,
           ACT_LEAKY02, nullptr, stats, stream);
    k_permute_rows<<<gblk((long)BB * NN * 3), 256, 0, stream>>>(points, perm, coords0, BB, NN, 6, 3);
    k_permute_rows<<<gblk((long)BB * NN * 8), 256, 0, stream>>>(xStart, perm, xPerm, BB, NN, 8, 8);

    // ---- encoder ----
    const int cinL[4] = {8, 32, 128, 256};
    const int coL [4] = {16, 64, 128, 256};
    float* featIn = xPerm;
    float* featNext[2] = {featBuf0, featBuf1};
    int n = NN;
    for (int l = 0; l < 4; ++l) {
        const int cin = cinL[l], co = coL[l], c2 = co / 2;
        const long R2 = (long)BB * n;
        const long RK = R2 * KK;

        dim3 gknn((unsigned)((n + 127) / 128), BB);
        k_knn16<<<gknn, 128, 0, stream>>>(coords0, n, NN * 3, idxB, distB);

        // mlp1: feats -> co/2, leaky(0.2)
        gemm(featIn, L[l].m1w, L[l].m1b, x1, R2, cin, c2, ACT_LEAKY02, stream);

        // lse1
        k_build_rel<<<gblk(RK), 256, 0, stream>>>(coords0, NN * 3, idxB, distB, relB, n, BB);
        gemm(relB, L[l].l1w, L[l].l1b, encRaw, RK, 10, c2, ACT_NONE, stream);
        run_bn(encRaw, L[l].l1g, L[l].l1be, RK, c2, xBig, co, 0, ACT_RELU, nullptr, stats, stream);
        k_broadcast<<<gblk(R2 * KK * c2), 256, 0, stream>>>(x1, xBig, R2, c2, co, c2);

        // pool1
        gemm(xBig, L[l].p1s, nullptr, scoresRaw, RK, co, co, ACT_NONE, stream);
        k_attpool<<<gblk(R2 * co), 256, 0, stream>>>(scoresRaw, xBig, pooled, R2, co);
        gemm(pooled, L[l].p1mw, L[l].p1mb, poolRaw, R2, co, c2, ACT_NONE, stream);
        run_bn(poolRaw, L[l].p1mg, L[l].p1mbe, R2, c2, x1, c2, 0, ACT_RELU, nullptr, stats, stream);

        // lse2 (same idx/dist/rel geometry)
        gemm(relB, L[l].l2w, L[l].l2b, encRaw, RK, 10, c2, ACT_NONE, stream);
        run_bn(encRaw, L[l].l2g, L[l].l2be, RK, c2, xBig, co, 0, ACT_RELU, nullptr, stats, stream);
        k_broadcast<<<gblk(R2 * KK * c2), 256, 0, stream>>>(x1, xBig, R2, c2, co, c2);

        // pool2
        gemm(xBig, L[l].p2s, nullptr, scoresRaw, RK, co, co, ACT_NONE, stream);
        k_attpool<<<gblk(R2 * co), 256, 0, stream>>>(scoresRaw, xBig, pooled, R2, co);
        gemm(pooled, L[l].p2mw, L[l].p2mb, poolRaw, R2, co, co, ACT_NONE, stream);
        run_bn(poolRaw, L[l].p2mg, L[l].p2mbe, R2, co, pool2out, co, 0, ACT_RELU, nullptr, stats, stream);

        // mlp2 + bn(shortcut), leaky(0.01)
        gemm(pool2out, L[l].m2w, L[l].m2b, mlp2out, R2, co, 2 * co, ACT_NONE, stream);
        gemm(featIn, L[l].shw, L[l].shb, shortRaw, R2, cin, 2 * co, ACT_NONE, stream);
        run_bn(shortRaw, L[l].shg, L[l].shbe, R2, 2 * co, stacks[l], 2 * co, 0,
               ACT_LEAKY001, mlp2out, stats, stream);

        // decimate by 4
        int n2 = n / 4;
        float* nxt = featNext[l & 1];
        k_slice_rows<<<gblk((long)BB * n2 * 2 * co), 256, 0, stream>>>(stacks[l], nxt, BB, n, n2, 2 * co);
        featIn = nxt;
        n = n2;
    }

    // ---- mid: 512 -> 512, relu (n == 16) ----
    gemm(featIn, mid_w, mid_b, midX, (long)BB * 16, 512, 512, ACT_RELU, stream);

    // ---- decoder ----
    float* x = midX;
    int n_cur = 16, Cx = 512;
    const int CdL[4] = {256, 128, 32, 8};
    const int CsL[4] = {512, 256, 128, 32};
    float* decB[2] = {decBufA, decBufB};
    for (int i = 0; i < 4; ++i) {
        int n_up = n_cur * 4;
        int Cs = CsL[i], Cd = CdL[i], Ccat = Cx + Cs;
        dim3 gnn((unsigned)((n_up + 127) / 128), BB);
        k_nn1<<<gnn, 128, 0, stream>>>(coords0, n_up, n_cur, NN * 3, nbIdx);
        k_gather_rows<<<gblk((long)BB * n_up * Cx), 256, 0, stream>>>(x, nbIdx, catBuf, BB, n_up, n_cur, Cx, Ccat, 0);
        k_copy_cols<<<gblk((long)BB * n_up * Cs), 256, 0, stream>>>(stacks[3 - i], catBuf, (long)BB * n_up, Cs, Ccat, Cx);
        gemm(catBuf, Dp[i].w, Dp[i].b, decRaw, (long)BB * n_up, Ccat, Cd, ACT_NONE, stream);
        run_bn(decRaw, Dp[i].g, Dp[i].be, (long long)BB * n_up, Cd, decB[i & 1], Cd, 0,
               ACT_RELU, nullptr, stats, stream);
        x = decB[i & 1]; Cx = Cd; n_cur = n_up;
    }

    // ---- un-permute, fc_end, transpose ----
    k_invperm<<<(NN + 255) / 256, 256, 0, stream>>>(perm, invp, NN);
    k_permute_rows<<<gblk((long)BB * NN * 8), 256, 0, stream>>>(x, invp, unperm, BB, NN, 8, 8);

    gemm(unperm, Fe[0].w, Fe[0].b, fc1raw, (long)BB * NN, 8, 64, ACT_NONE, stream);
    run_bn(fc1raw, Fe[0].g, Fe[0].be, (long long)BB * NN, 64, fc1, 64, 0, ACT_RELU, nullptr, stats, stream);
    gemm(fc1, Fe[1].w, Fe[1].b, fc2raw, (long)BB * NN, 64, 32, ACT_NONE, stream);
    run_bn(fc2raw, Fe[1].g, Fe[1].be, (long long)BB * NN, 32, fc2, 32, 0, ACT_RELU, nullptr, stats, stream);
    gemm(fc2, Fe[2].w, Fe[2].b, logits, (long)BB * NN, 32, 13, ACT_NONE, stream);

    k_transpose_out<<<gblk((long)BB * NN * 13), 256, 0, stream>>>(logits, (float*)d_out, BB, NN, 13);
}
